// Attention_86629490360504
// MI455X (gfx1250) — matrix operand order repaired and checked by reading, not compiled, not run on hardware
//
#include <hip/hip_runtime.h>

typedef __attribute__((ext_vector_type(16))) _Float16 v16h;
typedef __attribute__((ext_vector_type(8)))  _Float16 v8h;
typedef __attribute__((ext_vector_type(4)))  _Float16 v4h;
typedef __attribute__((ext_vector_type(8)))  float    v8f;

union U16h { v16h v; v8h h[2]; _Float16 e[16]; };
union U8f  { v8f v; float e[8]; };

#define NB 128
#define LQ 44
#define CW 1024
#define LWIN 704

#define QS_STRIDE 1032              // halves, padded vs 1024 for bank spread
#define KS_BLK    72                // halves per 64-channel block (padded)
#define KS_STRIDE (16*KS_BLK)       // 1152 halves per key row
#define PS_STRIDE 712               // halves
#define VT_STRIDE 712               // halves
#define S_STRIDE  712               // floats

#define QS_BYTES (16*QS_STRIDE*2)   // 33024
#define KS_BYTES (LQ*KS_STRIDE*2)   // 101376
#define S_BYTES  (16*S_STRIDE*4)    // 45568
#define SMEM_BYTES (QS_BYTES + KS_BYTES + S_BYTES)  // 179968 < 320KB

__global__ __launch_bounds__(256)
void sr_attn_kernel(const float* __restrict__ q, const float* __restrict__ kmat,
                    const float* __restrict__ vmat, const float* __restrict__ rel,
                    float* __restrict__ out_val, float* __restrict__ out_attn)
{
    __shared__ __align__(16) char smem[SMEM_BYTES];
    _Float16* Qs  = (_Float16*)smem;                         // 16 x QS_STRIDE (phase A)
    _Float16* Ps  = (_Float16*)smem;                         // 16 x PS_STRIDE (phase B, reuse)
    _Float16* Ks  = (_Float16*)(smem + QS_BYTES);            // 44 x KS_STRIDE (phase A)
    _Float16* VsT = (_Float16*)(smem + QS_BYTES);            // 64 x VT_STRIDE (phase B, reuse)
    float*    S   = (float*)(smem + QS_BYTES + KS_BYTES);    // 16 x S_STRIDE

    const int b    = blockIdx.x;
    const int lt   = blockIdx.y;      // query row tile: rows [lt*16, lt*16+16)
    const int row0 = lt * 16;
    const int tid  = threadIdx.x;
    const int wave = tid >> 5;
    const int lane = tid & 31;
    const int hi   = lane >> 4;       // 0: lanes 0-15, 1: lanes 16-31
    const int n    = lane & 15;       // A-row / B-col within tile

    // ---------- stage Q tile (f32 -> f16), zero-pad rows >= 44 ----------
    for (int i = tid; i < 16 * (CW/4); i += 256) {
        int r  = i >> 8;              // 256 float4 per row
        int c4 = (i & 255) << 2;
        int grow = row0 + r;
        float4 f = make_float4(0.f, 0.f, 0.f, 0.f);
        if (grow < LQ) f = ((const float4*)(q + ((size_t)b*LQ + grow)*CW))[c4 >> 2];
        v4h hv; hv.x=(_Float16)f.x; hv.y=(_Float16)f.y; hv.z=(_Float16)f.z; hv.w=(_Float16)f.w;
        *(v4h*)&Qs[r*QS_STRIDE + c4] = hv;
    }
    // ---------- stage K (f32 -> f16), 64-block padded layout ----------
    for (int i = tid; i < LQ * (CW/4); i += 256) {
        int r  = i >> 8;
        int c4 = (i & 255) << 2;
        float4 f = ((const float4*)(kmat + ((size_t)b*LQ + r)*CW))[c4 >> 2];
        int blk = c4 >> 6, off = c4 & 63;
        v4h hv; hv.x=(_Float16)f.x; hv.y=(_Float16)f.y; hv.z=(_Float16)f.z; hv.w=(_Float16)f.w;
        *(v4h*)&Ks[r*KS_STRIDE + blk*KS_BLK + off] = hv;
    }
    __syncthreads();

    // ---------- phase A: scores = Q * K_g^T (mesh-gather folded into addressing) ----------
    const int i_ = n >> 2, k_ = n & 3;     // (i,k) of m-column n = i*4+k
    const float awA = (i_ == 0) ? 1.0f : ((i_ == 2) ? 0.5f : 0.75f);
    const float awB = (k_ == 0) ? 1.0f : ((k_ == 2) ? 0.5f : 0.75f);
    const float mval = awA * awB - 1.0f;   // _sr_attn_mask value for this column

    for (int mt = wave; mt < LQ; mt += 8) {         // one key row per m-tile
        v8f acc = {};
        const _Float16* arow = &Qs[n * QS_STRIDE];
        const _Float16* krow = &Ks[mt * KS_STRIDE];
        #pragma unroll 4
        for (int s = 0; s < 32; ++s) {              // K = 1024 in steps of 32
            U16h A, Bm;
            int akb = s*32 + (hi << 4);             // A: K = base..base+15
            A.h[0] = *(const v8h*)(arow + akb);
            A.h[1] = *(const v8h*)(arow + akb + 8);
            int cstart = s*32 + (hi << 4);          // B: 16 consecutive dest channels
            int cb = cstart >> 6, coff = cstart & 63;
            int j_ = cb >> 2, l_ = cb & 3;
            int sb = (((j_ - i_) & 3) << 2) | ((l_ - k_) & 3);  // source 64-block
            const _Float16* bp = krow + sb*KS_BLK + coff;
            Bm.h[0] = *(const v8h*)bp;
            Bm.h[1] = *(const v8h*)(bp + 8);
            acc = __builtin_amdgcn_wmma_f32_16x16x32_f16(
                      false, A.v, false, Bm.v, (short)0, acc, false, false);
        }
        U8f c; c.v = acc;
        int m = mt*16 + n;
        #pragma unroll
        for (int r = 0; r < 8; ++r) {
            int rloc = r + (hi << 3);
            int grow = row0 + rloc;
            if (grow < LQ) {
                float sc = (c.e[r] + mval) * 0.0625f + rel[grow*LWIN + m];
                S[rloc*S_STRIDE + m] = sc;
            }
        }
    }
    __syncthreads();

    // ---------- softmax per row (one wave per row) + p_attn output + f16 P ----------
    #pragma unroll
    for (int rr = 0; rr < 2; ++rr) {
        int rloc = wave*2 + rr;
        int grow = row0 + rloc;
        if (grow < LQ) {
            float vals[22];
            float mx = -3.0e38f;
            #pragma unroll
            for (int j2 = 0; j2 < 22; ++j2) {
                vals[j2] = S[rloc*S_STRIDE + j2*32 + lane];
                mx = fmaxf(mx, vals[j2]);
            }
            #pragma unroll
            for (int o = 16; o >= 1; o >>= 1) mx = fmaxf(mx, __shfl_xor(mx, o, 32));
            float sum = 0.f;
            #pragma unroll
            for (int j2 = 0; j2 < 22; ++j2) { vals[j2] = __expf(vals[j2] - mx); sum += vals[j2]; }
            #pragma unroll
            for (int o = 16; o >= 1; o >>= 1) sum += __shfl_xor(sum, o, 32);
            float inv = 1.0f / sum;
            float* pa = out_attn + ((size_t)b*LQ + grow)*LWIN;
            #pragma unroll
            for (int j2 = 0; j2 < 22; ++j2) {
                float p = vals[j2] * inv;
                pa[j2*32 + lane] = p;
                Ps[rloc*PS_STRIDE + j2*32 + lane] = (_Float16)p;
            }
        } else {
            #pragma unroll
            for (int j2 = 0; j2 < 22; ++j2) Ps[rloc*PS_STRIDE + j2*32 + lane] = (_Float16)0.f;
        }
    }
    // ---------- stage V transposed: VsT[cc][m'] = value[b, m'/16, (m'%16)*64 + cc] ----------
    for (int i = tid; i < LQ * (CW/4); i += 256) {
        int r  = i >> 8;
        int c4 = (i & 255) << 2;
        float4 f = ((const float4*)(vmat + ((size_t)b*LQ + r)*CW))[c4 >> 2];
        int blk = c4 >> 6, cc = c4 & 63;
        int mp = r*16 + blk;
        VsT[(cc+0)*VT_STRIDE + mp] = (_Float16)f.x;
        VsT[(cc+1)*VT_STRIDE + mp] = (_Float16)f.y;
        VsT[(cc+2)*VT_STRIDE + mp] = (_Float16)f.z;
        VsT[(cc+3)*VT_STRIDE + mp] = (_Float16)f.w;
    }
    __syncthreads();

    // ---------- phase B: p_val[l, cb*64+cc] = sum_m' Pperm_cb[l,m'] * Vr[m',cc] ----------
    for (int cbt = wave; cbt < 16; cbt += 8) {      // output 64-channel block
        int jb = cbt >> 2, lb = cbt & 3;
        v8f acc[4] = {};
        for (int s = 0; s < 22; ++s) {              // m' = 704 in steps of 32
            U16h A;
            int base = s*32 + (hi << 4);
            #pragma unroll
            for (int e = 0; e < 16; ++e) {          // column-permuted P gather
                int mp = base + e;
                int lidx = mp >> 4, w16 = mp & 15;
                int ip = w16 >> 2, kp = w16 & 3;
                int src = (((jb - ip) & 3) << 2) | ((lb - kp) & 3);
                A.e[e] = Ps[n*PS_STRIDE + lidx*16 + src];
            }
            int mb = s*32 + (hi << 4);
            #pragma unroll
            for (int cct = 0; cct < 4; ++cct) {     // 4 WMMAs share one A gather
                U16h Bm;
                const _Float16* bp = &VsT[(cct*16 + n)*VT_STRIDE + mb];
                Bm.h[0] = *(const v8h*)bp;
                Bm.h[1] = *(const v8h*)(bp + 8);
                acc[cct] = __builtin_amdgcn_wmma_f32_16x16x32_f16(
                               false, A.v, false, Bm.v, (short)0, acc[cct], false, false);
            }
        }
        #pragma unroll
        for (int cct = 0; cct < 4; ++cct) {
            U8f c; c.v = acc[cct];
            #pragma unroll
            for (int r = 0; r < 8; ++r) {
                int rloc = r + (hi << 3);
                int grow = row0 + rloc;
                if (grow < LQ)
                    out_val[((size_t)b*LQ + grow)*CW + cbt*64 + cct*16 + n] = c.e[r];
            }
        }
    }
}

extern "C" void kernel_launch(void* const* d_in, const int* in_sizes, int n_in,
                              void* d_out, int out_size, void* d_ws, size_t ws_size,
                              hipStream_t stream) {
    const float* q   = (const float*)d_in[0];
    const float* k   = (const float*)d_in[1];
    const float* v   = (const float*)d_in[2];
    // d_in[3] = window_size (int, fixed at 4 in this problem)
    const float* rel = (const float*)d_in[4];
    float* out_val  = (float*)d_out;
    float* out_attn = out_val + (size_t)NB*LQ*CW;   // tuple: (p_val, p_attn) concatenated
    dim3 grid(NB, 3);
    sr_attn_kernel<<<grid, 256, 0, stream>>>(q, k, v, rel, out_val, out_attn);
}
